// LSTM_14439680049244
// MI455X (gfx1250) — matrix operand order repaired and checked by reading, not compiled, not run on hardware
//
#include <hip/hip_runtime.h>
#include <hip/hip_bf16.h>

typedef __bf16 bf16_t;
typedef __attribute__((ext_vector_type(16))) __bf16 v16bf;
typedef __attribute__((ext_vector_type(8)))  __bf16 v8bf;
typedef __attribute__((ext_vector_type(8)))  float  v8f;

// ---------------------------------------------------------------------------
// WMMA operand loaders per CDNA5 ISA §7.12.2 (wave32 layouts)
// ---------------------------------------------------------------------------

// A matrix 16x32 bf16, row-major source with leading dim `lda` (elements).
// lanes 0-15: M=lane, K=0..7 (elems 0..7) and K=16..23 (elems 8..15)
// lanes 16-31: M=lane-16, K=8..15 and K=24..31
__device__ __forceinline__ v16bf load_a_bf16(const bf16_t* base, long lda) {
    int lane = threadIdx.x & 31;
    int row  = lane & 15;
    int kb   = (lane >> 4) << 3;   // 0 or 8
    const bf16_t* p = base + (long)row * lda + kb;
    v8bf lo = *(const v8bf*)(p);
    v8bf hi = *(const v8bf*)(p + 16);
    v16bf a;
#pragma unroll
    for (int i = 0; i < 8; ++i) { a[i] = lo[i]; a[i + 8] = hi[i]; }
    return a;
}

// B matrix 32x16 bf16 from a [N x K] source, ld = K (global memory path).
__device__ __forceinline__ v16bf load_b_bf16(const bf16_t* base, long ldb) {
    int lane = threadIdx.x & 31;
    int col  = lane & 15;
    int kb   = (lane >> 4) << 3;   // 0 or 8
    const bf16_t* p = base + (long)col * ldb + kb;
    v8bf lo = *(const v8bf*)(p);
    v8bf hi = *(const v8bf*)(p + 16);
    v16bf b;
#pragma unroll
    for (int i = 0; i < 8; ++i) { b[i] = lo[i]; b[i + 8] = hi[i]; }
    return b;
}

// Same B layout but sourced from LDS-staged [16 cols][1024 K] tile.
__device__ __forceinline__ v16bf load_b_lds(const bf16_t* sB, int k0) {
    int lane = threadIdx.x & 31;
    int col  = lane & 15;
    int kb   = (lane >> 4) << 3;
    const bf16_t* p = sB + col * 1024 + k0 + kb;
    v8bf lo = *(const v8bf*)(p);
    v8bf hi = *(const v8bf*)(p + 16);
    v16bf b;
#pragma unroll
    for (int i = 0; i < 8; ++i) { b[i] = lo[i]; b[i + 8] = hi[i]; }
    return b;
}

// ---------------------------------------------------------------------------
// Grid-wide barrier (sense-free, monotonic phase). Control words are zeroed
// by the preceding k_xproj launch, so every persistent launch starts clean.
// ---------------------------------------------------------------------------
__device__ __forceinline__ void grid_sync(int* cnt, int* phase, int nblk, int gen) {
    __threadfence();        // release: flush this wave's global stores
    __syncthreads();
    if (threadIdx.x == 0) {
        int prev = __hip_atomic_fetch_add(cnt, 1, __ATOMIC_ACQ_REL,
                                          __HIP_MEMORY_SCOPE_AGENT);
        if (prev == nblk - 1) {
            __hip_atomic_store(cnt, 0, __ATOMIC_RELAXED, __HIP_MEMORY_SCOPE_AGENT);
            __hip_atomic_store(phase, gen, __ATOMIC_RELEASE, __HIP_MEMORY_SCOPE_AGENT);
        } else {
            while (__hip_atomic_load(phase, __ATOMIC_ACQUIRE,
                                     __HIP_MEMORY_SCOPE_AGENT) < gen) {
                __builtin_amdgcn_s_sleep(1);
            }
        }
    }
    __syncthreads();
    __threadfence();        // acquire: invalidate so we see other blocks' h_t
}

// ---------------------------------------------------------------------------
// Prep kernels: transpose / convert fp32 -> bf16
// ---------------------------------------------------------------------------
__global__ void k_transpose_bf16(const float* __restrict__ in,
                                 bf16_t* __restrict__ out, int R, int C) {
    long i = (long)blockIdx.x * blockDim.x + threadIdx.x;
    long total = (long)R * C;
    if (i >= total) return;
    long r = i % R;
    long c = i / R;
    out[i] = (bf16_t)in[r * (long)C + c];
}

__global__ void k_convert_bf16(const float* __restrict__ in,
                               bf16_t* __restrict__ out, long n) {
    long i = (long)blockIdx.x * blockDim.x + threadIdx.x;
    if (i < n) out[i] = (bf16_t)in[i];
}

// ---------------------------------------------------------------------------
// xproj chunk GEMM: xpc[tc*32+m][g] = sum_d X[m][t0+tc][d] * Wx[d][g]
// M = 64*32 = 2048 rows, N = 4096, K = 512.  Also zeroes barrier control.
// ---------------------------------------------------------------------------
__global__ void __launch_bounds__(256)
k_xproj(const float* __restrict__ X, int t0,
        const bf16_t* __restrict__ WxT,   // [4096 x 512] bf16
        float* __restrict__ xpc,          // [2048 x 4096] f32
        int* __restrict__ ctrl) {
    if (blockIdx.x == 0 && threadIdx.x < 2) ctrl[threadIdx.x] = 0;

    const int Dk = 512, G = 4096, T = 512;
    int wave = threadIdx.x >> 5, lane = threadIdx.x & 31;
    int tile  = blockIdx.x * 8 + wave;    // 32768 tiles
    int ntile = tile & 255;               // 256 n-tiles
    int mtile = tile >> 8;                // 128 m-tiles
    int n0 = ntile * 16, r0 = mtile * 16;

    int r  = r0 + (lane & 15);            // row within chunk: r = tc*32 + m
    int m  = r & 31;
    int tc = r >> 5;
    const float* arow = X + ((long)m * T + (t0 + tc)) * Dk;
    int kb = (lane >> 4) << 3;

    v8f acc = {};
    for (int k0 = 0; k0 < Dk; k0 += 32) {
        const float* p = arow + k0 + kb;
        v16bf a;
#pragma unroll
        for (int i = 0; i < 8; ++i) { a[i] = (bf16_t)p[i]; a[i + 8] = (bf16_t)p[16 + i]; }
        v16bf b = load_b_bf16(WxT + (long)n0 * Dk + k0, Dk);
        acc = __builtin_amdgcn_wmma_f32_16x16x32_bf16(false, a, false, b,
                                                      (short)0, acc, false, false);
    }
    int col = n0 + (lane & 15);
    int rb  = r0 + ((lane >> 4) << 3);
#pragma unroll
    for (int v = 0; v < 8; ++v)
        xpc[(long)(rb + v) * G + col] = acc[v];
}

// ---------------------------------------------------------------------------
// Persistent LSTM chunk: TC sequential timesteps in ONE launch, grid barrier
// between steps.  64 blocks x 128 threads (4 waves).
//   Block owns one 16-col n-tile of H (all 4 gates, both m-tiles).
//   Its Wh slice (4 gates x 16 cols x K=1024 bf16 = 128 KB) is staged into
//   LDS ONCE, so every step's B-operand feeds from LDS, not L2.
//   wave = K-quarter (K=256 each); 4-way LDS reduction; wave 0 ends up with
//   all 4 gates of the tile -> LSTM cell math stays in registers.
// Dynamic LDS: 128 KB staged Wh + 24 KB reduction = 152 KB (< 320 KB/WGP).
// ---------------------------------------------------------------------------
__global__ void __launch_bounds__(128)
k_step_chunk(const bf16_t* __restrict__ h0b,   // [32 x 1024] (t==0 only)
             bf16_t* __restrict__ hs,          // [32][512][1024] bf16
             const bf16_t* __restrict__ WhT,   // [4096 x 1024] bf16
             const float* __restrict__ xpc,    // [TC][32][4096] f32
             const float* __restrict__ bias,   // [4096]
             const float* __restrict__ c0,     // [32 x 1024] (t==0 only)
             float* __restrict__ cbuf,         // [32 x 1024]
             int t0, int TC,
             int* __restrict__ cnt, int* __restrict__ phase) {
    const int Hh = 1024;
    const long ldHS = 512L * 1024;
    int wave = threadIdx.x >> 5, lane = threadIdx.x & 31;
    int n0   = blockIdx.x * 16;             // global column base, 0..1008
    int kbeg = wave * 256;                  // K-quarter

    extern __shared__ bf16_t smem[];
    bf16_t* sWh = smem;                               // [4][16][1024] bf16
    float*  red = (float*)(smem + 4 * 16 * 1024);     // [3][8][32][8]  f32

    // ---- stage this block's Wh slice into LDS (once per launch) ----
    {
        uint4* dst = (uint4*)sWh;
        for (int u = threadIdx.x; u < 8192; u += 128) {   // 8192 x 16B = 128 KB
            int k8 = u & 127;            // 8-element chunk along K
            int c  = (u >> 7) & 15;      // column within tile
            int g  = u >> 11;            // gate
            const uint4* src =
                (const uint4*)(WhT + ((long)(g * Hh + n0 + c) * Hh + k8 * 8));
            dst[u] = *src;
        }
    }
    __syncthreads();

    for (int tc = 0; tc < TC; ++tc) {
        int t = t0 + tc;
        const bf16_t* hPrev;
        long ldh;
        const float* cIn;
        if (t == 0) { hPrev = h0b; ldh = Hh; cIn = c0; }
        else        { hPrev = hs + (long)(t - 1) * Hh; ldh = ldHS; cIn = cbuf; }
        const float* xp = xpc + (long)tc * 32 * 4096;

        v8f acc[2][4] = {};
        for (int k0 = kbeg; k0 < kbeg + 256; k0 += 32) {
            v16bf a0 = load_a_bf16(hPrev + k0, ldh);
            v16bf a1 = load_a_bf16(hPrev + 16 * ldh + k0, ldh);
#pragma unroll
            for (int g = 0; g < 4; ++g) {
                v16bf bm = load_b_lds(sWh + g * 16 * 1024, k0);
                acc[0][g] = __builtin_amdgcn_wmma_f32_16x16x32_bf16(
                    false, a0, false, bm, (short)0, acc[0][g], false, false);
                acc[1][g] = __builtin_amdgcn_wmma_f32_16x16x32_bf16(
                    false, a1, false, bm, (short)0, acc[1][g], false, false);
            }
        }

        // K-split reduction: waves 1..3 dump partials to LDS
        if (wave != 0) {
#pragma unroll
            for (int mt = 0; mt < 2; ++mt)
#pragma unroll
                for (int g = 0; g < 4; ++g)
                    *(v8f*)&red[(((wave - 1) * 8 + mt * 4 + g) * 32 + lane) * 8]
                        = acc[mt][g];
        }
        __syncthreads();

        if (wave == 0) {
#pragma unroll
            for (int mt = 0; mt < 2; ++mt)
#pragma unroll
                for (int g = 0; g < 4; ++g)
#pragma unroll
                    for (int w = 0; w < 3; ++w)
                        acc[mt][g] +=
                            *(const v8f*)&red[((w * 8 + mt * 4 + g) * 32 + lane) * 8];

            int n = n0 + (lane & 15);
            float bi = bias[n],            bf2 = bias[Hh + n];
            float bo = bias[2 * Hh + n],   bg  = bias[3 * Hh + n];
#pragma unroll
            for (int mt = 0; mt < 2; ++mt) {
                int mb = mt * 16 + ((lane >> 4) << 3);
#pragma unroll
                for (int v = 0; v < 8; ++v) {
                    int m = mb + v;
                    const float* xr = xp + (long)m * 4096;
                    float gi = xr[n]           + bi  + acc[mt][0][v];
                    float gf = xr[Hh + n]      + bf2 + acc[mt][1][v];
                    float go = xr[2 * Hh + n]  + bo  + acc[mt][2][v];
                    float gg = xr[3 * Hh + n]  + bg  + acc[mt][3][v];
                    float c  = cIn[(long)m * Hh + n];
                    float cn = c * gf + gi * gg;
                    float hn = go * tanhf(cn);
                    cbuf[(long)m * Hh + n]                = cn;
                    hs[(long)m * ldHS + (long)t * Hh + n] = (bf16_t)hn;
                }
            }
        }

        // publish h_t / protect LDS reuse; not needed after final step
        if (tc != TC - 1) grid_sync(cnt, phase, 64, tc + 1);
    }
}

// ---------------------------------------------------------------------------
// Output GEMM: out[r][o] = sum_h hs[r][h] * W_fc[o][h] + b_fc[o]
// M = 16384, N = 512, K = 1024. 8 waves/block.
// ---------------------------------------------------------------------------
__global__ void __launch_bounds__(256)
k_out(const bf16_t* __restrict__ hs,      // [16384 x 1024] bf16
      const bf16_t* __restrict__ Wfc,     // [512 x 1024] bf16 (B-layout)
      const float* __restrict__ bfc,
      float* __restrict__ out) {
    const int Hh = 1024, O = 512;
    int wave = threadIdx.x >> 5, lane = threadIdx.x & 31;
    int tile  = blockIdx.x * 8 + wave;    // 32768 tiles
    int ntile = tile & 31;                // 32 n-tiles
    int mtile = tile >> 5;                // 1024 m-tiles
    int n0 = ntile * 16, r0 = mtile * 16;

    v8f acc = {};
    for (int k0 = 0; k0 < Hh; k0 += 32) {
        v16bf a = load_a_bf16(hs + (long)r0 * Hh + k0, Hh);
        v16bf b = load_b_bf16(Wfc + (long)n0 * Hh + k0, Hh);
        acc = __builtin_amdgcn_wmma_f32_16x16x32_bf16(false, a, false, b,
                                                      (short)0, acc, false, false);
    }
    int o  = n0 + (lane & 15);
    float bb = bfc[o];
    int rb = r0 + ((lane >> 4) << 3);
#pragma unroll
    for (int v = 0; v < 8; ++v)
        out[(long)(rb + v) * O + o] = acc[v] + bb;
}

// ---------------------------------------------------------------------------
// Host driver
// ---------------------------------------------------------------------------
extern "C" void kernel_launch(void* const* d_in, const int* in_sizes, int n_in,
                              void* d_out, int out_size, void* d_ws, size_t ws_size,
                              hipStream_t stream) {
    (void)in_sizes; (void)n_in; (void)out_size; (void)ws_size;
    const float* X   = (const float*)d_in[0];
    const float* h0  = (const float*)d_in[1];
    const float* c0  = (const float*)d_in[2];
    const float* Wx  = (const float*)d_in[3];
    const float* Wh  = (const float*)d_in[4];
    const float* b   = (const float*)d_in[5];
    const float* Wfc = (const float*)d_in[6];
    const float* bfc = (const float*)d_in[7];
    float* out = (float*)d_out;

    const int N = 32, T = 512, D = 512, Hh = 1024, O = 512, G = 4096, TC = 64;
    const size_t SMEM_STEP = 4 * 16 * 1024 * 2   // staged Wh slice (128 KB)
                           + 3 * 8 * 32 * 8 * 4; // reduction scratch (24 KB)

    // Opt in to >64KB dynamic LDS (CDNA5 WGP has 320 KB). Host-side, idempotent.
    (void)hipFuncSetAttribute((const void*)k_step_chunk,
                              hipFuncAttributeMaxDynamicSharedMemorySize,
                              (int)SMEM_STEP);

    char* ws = (char*)d_ws;
    size_t off = 0;
    auto alloc = [&](size_t bytes) {
        void* p = ws + off;
        off = (off + bytes + 255) & ~(size_t)255;
        return p;
    };
    bf16_t* WxT  = (bf16_t*)alloc((size_t)G * D * 2);       //  4 MB
    bf16_t* WhT  = (bf16_t*)alloc((size_t)G * Hh * 2);      //  8 MB
    bf16_t* Wfb  = (bf16_t*)alloc((size_t)O * Hh * 2);      //  1 MB
    bf16_t* h0b  = (bf16_t*)alloc((size_t)N * Hh * 2);      // 64 KB
    float*  cbuf = (float*) alloc((size_t)N * Hh * 4);      // 128 KB
    int*    ctrl = (int*)   alloc(256);                     // barrier cnt/phase
    bf16_t* hs   = (bf16_t*)alloc((size_t)N * T * Hh * 2);  // 32 MB
    float*  xpc  = (float*) alloc((size_t)TC * N * G * 4);  // 32 MB
    // total ~81 MB of workspace

    // Prep: transposes/conversions to bf16
    { long tot = (long)G * D;
      k_transpose_bf16<<<(tot + 255) / 256, 256, 0, stream>>>(Wx, WxT, D, G); }
    { long tot = (long)G * Hh;
      k_transpose_bf16<<<(tot + 255) / 256, 256, 0, stream>>>(Wh, WhT, Hh, G); }
    { long tot = (long)O * Hh;
      k_convert_bf16<<<(tot + 255) / 256, 256, 0, stream>>>(Wfc, Wfb, tot); }
    { long tot = (long)N * Hh;
      k_convert_bf16<<<(tot + 255) / 256, 256, 0, stream>>>(h0, h0b, tot); }

    // Time-chunked recurrence: 1 xproj GEMM + 1 persistent 64-step kernel
    for (int chunk = 0; chunk < T / TC; ++chunk) {
        int t0 = chunk * TC;
        k_xproj<<<4096, 256, 0, stream>>>(X, t0, WxT, xpc, ctrl);
        k_step_chunk<<<64, 128, SMEM_STEP, stream>>>(h0b, hs, WhT, xpc, b, c0,
                                                     cbuf, t0, TC,
                                                     ctrl, ctrl + 1);
    }

    // Output projection
    k_out<<<4096, 256, 0, stream>>>(hs, Wfb, bfc, out);
}
